// ChannelAttention_21500606284471
// MI455X (gfx1250) — matrix operand order repaired and checked by reading, not compiled, not run on hardware
//
#include <hip/hip_runtime.h>
#include <hip/hip_bf16.h>
#include <stdint.h>

// ---------------------------------------------------------------------------
// Channel attention, B=32 C=1280 S=16 (HW=256).
// Compute-bound (~161 GFLOP vs ~110MB traffic -> 1.4 kFLOP/B): all six GEMMs
// run on v_wmma_f32_16x16x32_bf16 (f32 accumulate).
// Round-3: prefetch via GLOBAL_LOAD_ASYNC_TO_LDS_B128 (ASYNCcnt) so the next
// K-tile's HBM traffic overlaps compute instead of serializing through the
// in-order DScnt that a ds_store staging path would impose.
// ---------------------------------------------------------------------------

#define CC  1280
#define HW  256
#define NB  32

typedef __attribute__((ext_vector_type(8)))  float  v8f;
typedef __attribute__((ext_vector_type(16))) __bf16 v16bf;

union Frag16 { v16bf v; uint4 q[2]; };

__device__ __forceinline__ unsigned short f2bf(float x) {
    union { float f; unsigned int u; } t; t.f = x;
    unsigned int u = t.u;
    unsigned int r = u + 0x7FFFu + ((u >> 16) & 1u);   // round-to-nearest-even
    return (unsigned short)(r >> 16);
}

// async 16B global -> LDS copy, tracked by ASYNCcnt (not LOADcnt/DScnt).
// lds_addr: wave-relative LDS byte address (= low 32 bits of flat pointer);
// base: uniform 64-bit global base (SGPR pair); off: per-lane byte offset.
__device__ __forceinline__ void async_copy16(unsigned lds_addr, const void* base, unsigned off) {
    asm volatile("global_load_async_to_lds_b128 %0, %1, %2"
                 :: "v"(lds_addr), "v"(off), "s"(base) : "memory");
}
__device__ __forceinline__ void wait_async0() {
    asm volatile("s_wait_asynccnt 0x0" ::: "memory");
}

// ---------------------------------------------------------------------------
// x[b][c][n] f32  ->  Xt[b][n][c] bf16   (LDS tile transpose, coalesced I/O)
// ---------------------------------------------------------------------------
__global__ void k_transpose_x(const float* __restrict__ x, unsigned short* __restrict__ Xt) {
    __shared__ float tile[32][33];
    const int b  = blockIdx.z;
    const int nx = blockIdx.x * 32, cy = blockIdx.y * 32;
    const int tx = threadIdx.x, ty = threadIdx.y;      // block (32,8)
    const float* xb = x + (size_t)b * CC * HW;
#pragma unroll
    for (int i = 0; i < 4; ++i) {
        int c = cy + ty + i * 8;
        tile[ty + i * 8][tx] = xb[(size_t)c * HW + nx + tx];
    }
    __syncthreads();
    unsigned short* Xtb = Xt + (size_t)b * HW * CC;
#pragma unroll
    for (int i = 0; i < 4; ++i) {
        int n = nx + ty + i * 8;
        Xtb[(size_t)n * CC + cy + tx] = f2bf(tile[tx][ty + i * 8]);
    }
}

// ---------------------------------------------------------------------------
// flat f32 -> bf16 convert (weights)
// ---------------------------------------------------------------------------
__global__ void k_convert_bf16(const float* __restrict__ in, unsigned short* __restrict__ out, int n) {
    int i = blockIdx.x * blockDim.x + threadIdx.x;
    if (i < n) out[i] = f2bf(in[i]);
}

// ---------------------------------------------------------------------------
// Softmax over the BATCH axis (faithful nn.Softmax2d-on-3D quirk):
// attn[b,c,d] = exp(s[b,c,d]-max_b) / sum_b exp(...)   one thread per (c,d).
// ---------------------------------------------------------------------------
__global__ void k_softmax_batch(const float* __restrict__ scores, unsigned short* __restrict__ attn) {
    const int cd = blockIdx.x * blockDim.x + threadIdx.x;   // < CC*CC
    const size_t stride = (size_t)CC * CC;
    float v[NB];
    float mx = -3.4e38f;
#pragma unroll
    for (int b = 0; b < NB; ++b) {
        v[b] = scores[(size_t)b * stride + cd];
        mx = fmaxf(mx, v[b]);
    }
    float s = 0.f;
#pragma unroll
    for (int b = 0; b < NB; ++b) { v[b] = __expf(v[b] - mx); s += v[b]; }
    float inv = 1.0f / s;
#pragma unroll
    for (int b = 0; b < NB; ++b) attn[(size_t)b * stride + cd] = f2bf(v[b] * inv);
}

// ---------------------------------------------------------------------------
// NT GEMM:  C[m,n] = sum_k A[m,k]*B[n,k]   (A,B bf16 row-major, f32 acc)
// Block tile 128x256, K-step 32; 256 threads = 8 waves (2 M x 4 N), each wave
// computes 64x64 via 4x4 v_wmma_f32_16x16x32_bf16 (16 wmma : 16 ds_load_b128).
// Double-buffered LDS (61 KB of 320 KB WGP pool); tile k+1 is fetched with
// global_load_async_to_lds_b128 (ASYNCcnt) while tile k is computed; one
// s_wait_asynccnt + one barrier per K-step.
// LDS rows padded to 40 halves (80B = 20 banks): the 16-lane b128 fragment
// reads hit all 16 distinct 4-bank groups -> conflict-free.
// MODE 0: bf16 row-major out, +bias[m]
// MODE 1: bf16 TRANSPOSED out (out[n*ldC+m]), +bias[m] if non-null
// MODE 2: f32 row-major out (scores)
// MODE 3: f32 row-major out = alpha*(acc+bias[m]) + Xres[m*ldC+n]  (final)
// ---------------------------------------------------------------------------
#define BM 128
#define BN 256
#define BK 32
#define LP 40   // padded LDS row stride in halves

template <int MODE>
__global__ __launch_bounds__(256)
void k_gemm_nt(const unsigned short* __restrict__ A, size_t sA,
               const unsigned short* __restrict__ Bt, size_t sB,
               void* __restrict__ Cout, size_t sC,
               const float* __restrict__ bias,
               const float* __restrict__ Xres, size_t sX,
               const float* __restrict__ alphap,
               int ldA, int ldB, int ldC, int Kdim)
{
    __shared__ __align__(16) unsigned short As[2][BM * LP];
    __shared__ __align__(16) unsigned short Bs[2][BN * LP];

    const int tid  = threadIdx.x;
    const int bz   = blockIdx.z;
    const unsigned short* Ab = A  + (size_t)bz * sA;   // uniform -> SGPRs
    const unsigned short* Bb = Bt + (size_t)bz * sB;   // uniform -> SGPRs
    const int m0 = blockIdx.y * BM;
    const int n0 = blockIdx.x * BN;

    const int wave = tid >> 5, lane = tid & 31;
    const int wm = (wave & 1) * 64;     // 2 waves along M
    const int wn = (wave >> 1) * 64;    // 4 waves along N
    const int r  = lane & 15;           // lane row/col within 16
    const int hh = lane >> 4;           // half-wave select

    v8f zero = {0.f, 0.f, 0.f, 0.f, 0.f, 0.f, 0.f, 0.f};
    v8f acc[4][4];
#pragma unroll
    for (int i = 0; i < 4; ++i)
#pragma unroll
        for (int j = 0; j < 4; ++j) acc[i][j] = zero;

    // issue async global->LDS copies for one K-tile into buffer `buf`
    auto stage = [&](int k0, int buf) {
        // A tile: 128 rows x 32 halves = 512 16B-chunks, 2 per thread
#pragma unroll
        for (int t = 0; t < 2; ++t) {
            int chunk = tid + 256 * t;
            int row = chunk >> 2, kc = (chunk & 3) << 3;
            unsigned lds = (unsigned)(uintptr_t)&As[buf][row * LP + kc];
            unsigned off = (unsigned)(((unsigned)(m0 + row) * (unsigned)ldA + k0 + kc) * 2u);
            async_copy16(lds, Ab, off);
        }
        // B tile: 256 rows x 32 halves = 1024 chunks, 4 per thread
#pragma unroll
        for (int t = 0; t < 4; ++t) {
            int chunk = tid + 256 * t;
            int row = chunk >> 2, kc = (chunk & 3) << 3;
            unsigned lds = (unsigned)(uintptr_t)&Bs[buf][row * LP + kc];
            unsigned off = (unsigned)(((unsigned)(n0 + row) * (unsigned)ldB + k0 + kc) * 2u);
            async_copy16(lds, Bb, off);
        }
    };

    const int nk = Kdim / BK;
    stage(0, 0);

    for (int kt = 0; kt < nk; ++kt) {
        wait_async0();                        // this wave's tile-kt copies done
        __syncthreads();                      // every wave's copies visible;
                                              // also: all reads of buf^1 done
        const int buf = kt & 1;
        if (kt + 1 < nk) stage((kt + 1) * BK, buf ^ 1);   // overlaps compute

        // A frags (ISA 16-bit A 16x32): lane holds row m; halves 0-7 = k 8*hh..,
        // halves 8-15 = k 16+8*hh..
        Frag16 af[4];
#pragma unroll
        for (int i = 0; i < 4; ++i) {
            int row = wm + 16 * i + r;
            af[i].q[0] = *reinterpret_cast<const uint4*>(&As[buf][row * LP + 8 * hh]);
            af[i].q[1] = *reinterpret_cast<const uint4*>(&As[buf][row * LP + 16 + 8 * hh]);
        }
        // per N-column: one B frag (lane holds col n, k 8*hh.. and 16+8*hh..),
        // then 4 WMMAs against the resident A frags
#pragma unroll
        for (int j = 0; j < 4; ++j) {
            Frag16 bfr;
            int row = wn + 16 * j + r;
            bfr.q[0] = *reinterpret_cast<const uint4*>(&Bs[buf][row * LP + 8 * hh]);
            bfr.q[1] = *reinterpret_cast<const uint4*>(&Bs[buf][row * LP + 16 + 8 * hh]);
#pragma unroll
            for (int i = 0; i < 4; ++i)
                acc[i][j] = __builtin_amdgcn_wmma_f32_16x16x32_bf16(
                    false, af[i].v, false, bfr.v, (short)0, acc[i][j], false, false);
        }
    }

    // epilogue: C layout = VGPR v -> M = v + 8*hh, N = lane%16
#pragma unroll
    for (int i = 0; i < 4; ++i)
#pragma unroll
        for (int j = 0; j < 4; ++j) {
            int mbase = m0 + wm + 16 * i + 8 * hh;
            int n = n0 + wn + 16 * j + r;
#pragma unroll
            for (int v = 0; v < 8; ++v) {
                int m = mbase + v;
                float val = acc[i][j][v];
                if (MODE == 0) {
                    float bv = bias ? bias[m] : 0.f;
                    ((unsigned short*)Cout)[(size_t)bz * sC + (size_t)m * ldC + n] = f2bf(val + bv);
                } else if (MODE == 1) {
                    float bv = bias ? bias[m] : 0.f;
                    ((unsigned short*)Cout)[(size_t)bz * sC + (size_t)n * ldC + m] = f2bf(val + bv);
                } else if (MODE == 2) {
                    ((float*)Cout)[(size_t)bz * sC + (size_t)m * ldC + n] = val;
                } else {
                    float a = alphap[0];
                    float o = a * (val + bias[m]) + Xres[(size_t)bz * sX + (size_t)m * ldC + n];
                    ((float*)Cout)[(size_t)bz * sC + (size_t)m * ldC + n] = o;
                }
            }
        }
}

// ---------------------------------------------------------------------------
extern "C" void kernel_launch(void* const* d_in, const int* in_sizes, int n_in,
                              void* d_out, int out_size, void* d_ws, size_t ws_size,
                              hipStream_t stream) {
    const float* x     = (const float*)d_in[0];
    const float* Wq    = (const float*)d_in[1];
    const float* bq    = (const float*)d_in[2];
    const float* Wk    = (const float*)d_in[3];
    const float* bk    = (const float*)d_in[4];
    const float* Wv    = (const float*)d_in[5];
    const float* bv    = (const float*)d_in[6];
    const float* Wr    = (const float*)d_in[7];
    const float* br    = (const float*)d_in[8];
    const float* alpha = (const float*)d_in[9];

    // workspace carving (all multiples of 256B); total ~412 MB
    char* ws = (char*)d_ws;
    const size_t szXt  = (size_t)NB * HW * CC * 2;   // 20.97 MB
    const size_t szW   = (size_t)CC * CC * 2;        //  3.28 MB each
    const size_t szQK  = (size_t)NB * CC * HW * 2;   // 20.97 MB each
    const size_t szSc  = (size_t)NB * CC * CC * 4;   // 209.7 MB
    const size_t szAt  = (size_t)NB * CC * CC * 2;   // 104.9 MB

    unsigned short* Xt    = (unsigned short*)ws;                 ws += szXt;
    unsigned short* Wqb   = (unsigned short*)ws;                 ws += szW;
    unsigned short* Wkb   = (unsigned short*)ws;                 ws += szW;
    unsigned short* Wvb   = (unsigned short*)ws;                 ws += szW;
    unsigned short* Wrb   = (unsigned short*)ws;                 ws += szW;
    unsigned short* Qm    = (unsigned short*)ws;                 ws += szQK;  // [b][c][n]
    unsigned short* Km    = (unsigned short*)ws;                 ws += szQK;  // [b][c][n]
    unsigned short* Vt    = (unsigned short*)ws;                 ws += szQK;  // [b][n][c]
    unsigned short* attT  = (unsigned short*)ws;                 ws += szQK;  // [b][n][c]
    unsigned short* attnb = (unsigned short*)ws;                 ws += szAt;  // [b][c][d]
    float*          scor  = (float*)ws;                          ws += szSc;  // [b][c][d]

    const size_t strXt = (size_t)HW * CC;   // per-batch elems, Xt/Vt/attT
    const size_t strQK = (size_t)CC * HW;   // per-batch elems, Qm/Km/x/out
    const size_t strSc = (size_t)CC * CC;   // per-batch elems, scores/attn

    // 1) x -> Xt (bf16, transposed)
    k_transpose_x<<<dim3(HW / 32, CC / 32, NB), dim3(32, 8), 0, stream>>>(x, Xt);

    // 2) weights -> bf16
    {
        int n = CC * CC, blk = 256, grd = (n + blk - 1) / blk;
        k_convert_bf16<<<grd, blk, 0, stream>>>(Wq, Wqb, n);
        k_convert_bf16<<<grd, blk, 0, stream>>>(Wk, Wkb, n);
        k_convert_bf16<<<grd, blk, 0, stream>>>(Wv, Wvb, n);
        k_convert_bf16<<<grd, blk, 0, stream>>>(Wr, Wrb, n);
    }

    // 3) projections: M=C, N=HW, K=C ; A = W (shared), B = Xt[b]
    dim3 gProj(HW / BN, CC / BM, NB), blk256(256);
    k_gemm_nt<0><<<gProj, blk256, 0, stream>>>(Wqb, 0, Xt, strXt, Qm, strQK,
                                               bq, nullptr, 0, nullptr, CC, CC, HW, CC);
    k_gemm_nt<0><<<gProj, blk256, 0, stream>>>(Wkb, 0, Xt, strXt, Km, strQK,
                                               bk, nullptr, 0, nullptr, CC, CC, HW, CC);
    k_gemm_nt<1><<<gProj, blk256, 0, stream>>>(Wvb, 0, Xt, strXt, Vt, strXt,
                                               bv, nullptr, 0, nullptr, CC, CC, CC, CC);

    // 4) scores[b] = K[b] (CxHW) x Q[b]^T  -> f32 [C x C];  M=N=C, K=HW
    dim3 gSc(CC / BN, CC / BM, NB);
    k_gemm_nt<2><<<gSc, blk256, 0, stream>>>(Km, strQK, Qm, strQK, scor, strSc,
                                             nullptr, nullptr, 0, nullptr, HW, HW, CC, HW);

    // 5) softmax across batch axis
    k_softmax_batch<<<(CC * CC) / 256, 256, 0, stream>>>(scor, attnb);

    // 6) att[b] = attn[b] (CxC) x V[b]  -> stored transposed attT [n][c]
    k_gemm_nt<1><<<gProj, blk256, 0, stream>>>(attnb, strSc, Vt, strXt, attT, strXt,
                                               nullptr, nullptr, 0, nullptr, CC, CC, CC, CC);

    // 7) out = alpha * (Wr x att + br) + x   (f32, [b][c][n])
    k_gemm_nt<3><<<gProj, blk256, 0, stream>>>(Wrb, 0, attT, strXt, d_out, strQK,
                                               br, x, strQK, alpha, CC, CC, HW, CC);

    (void)in_sizes; (void)n_in; (void)out_size; (void)ws_size;
}
